// SparseAttentionWithGhostKV_1520418423123
// MI455X (gfx1250) — matrix operand order repaired and checked by reading, not compiled, not run on hardware
//
#include <hip/hip_runtime.h>
#include <hip/hip_bf16.h>
#include <math.h>

#define D_MODEL 1024
#define N_HEADS 16
#define HEAD_DIM 64
#define TOP_K   64
#define BATCH   2
#define SEQ     2048
#define M_ROWS  (BATCH*SEQ)   // 4096

typedef __attribute__((ext_vector_type(16))) __bf16 bf16x16;
typedef __attribute__((ext_vector_type(8)))  __bf16 bf16x8;
typedef __attribute__((ext_vector_type(8)))  float  f32x8;

union Frag { bf16x16 v; bf16x8 h[2]; };

__device__ inline unsigned short f2bf_bits(float f) {
  unsigned u; __builtin_memcpy(&u, &f, 4);
  unsigned r = u + 0x7FFFu + ((u >> 16) & 1u);
  return (unsigned short)(r >> 16);
}
__device__ inline __bf16 f2bf(float f) {
  unsigned short hs = f2bf_bits(f);
  __bf16 b; __builtin_memcpy(&b, &hs, 2); return b;
}
__device__ inline float bits2f(unsigned u) {
  float f; __builtin_memcpy(&f, &u, 4); return f;
}
// order-preserving float -> u32 map (larger float => larger uint)
__device__ inline unsigned f2ord(float f) {
  unsigned u; __builtin_memcpy(&u, &f, 4);
  return u ^ ((u >> 31) ? 0xFFFFFFFFu : 0x80000000u);
}
__device__ inline float ord2f(unsigned u) {
  u ^= ((u >> 31) ? 0x80000000u : 0xFFFFFFFFu);
  return bits2f(u);
}

__device__ inline f32x8 wmma_bf16(bf16x16 a, bf16x16 b, f32x8 c) {
  return __builtin_amdgcn_wmma_f32_16x16x32_bf16(
      false, a, false, b, (short)0, c, false, false);
}

// CDNA5 async global->LDS copy (ASYNCcnt tracked, bypasses VGPRs)
__device__ inline void async_ld_b128(unsigned lds_off, const __bf16* g) {
  asm volatile("global_load_async_to_lds_b128 %0, %1, off"
               :: "v"(lds_off), "v"((unsigned long long)(size_t)g) : "memory");
}
__device__ inline void wait_async0() {
  asm volatile("s_wait_asynccnt 0x0" ::: "memory");
}
__device__ inline unsigned lds_off32(const void* p) {  // generic ptr low 32 bits = LDS offset
  return (unsigned)(size_t)p;
}

// ---------------------------------------------------------------- convert
__global__ void cvt_f32_bf16_kernel(const float* __restrict__ in,
                                    __bf16* __restrict__ out, int n) {
  int i = blockIdx.x * blockDim.x + threadIdx.x;
  if (i < n) out[i] = f2bf(in[i]);
}

// ---------------------------------------------------- weight transpose f32->bf16
__global__ void transpose_w_kernel(const float* __restrict__ W,
                                   __bf16* __restrict__ WT) {
  __shared__ float t[32][33];
  int nx = blockIdx.x * 32 + threadIdx.x;
  for (int j = 0; j < 4; j++) {
    int k = blockIdx.y * 32 + threadIdx.y + j * 8;
    t[threadIdx.y + j * 8][threadIdx.x] = W[(size_t)k * D_MODEL + nx];
  }
  __syncthreads();
  int k2 = blockIdx.y * 32 + threadIdx.x;
  for (int j = 0; j < 4; j++) {
    int n2 = blockIdx.x * 32 + threadIdx.y + j * 8;
    WT[(size_t)n2 * D_MODEL + k2] = f2bf(t[threadIdx.x][threadIdx.y + j * 8]);
  }
}

// ---------------------------------------------------------------- GEMM core
// 256 threads = 8 waves (2x4). Each wave: 32x32 C block = 4 WMMA accumulators.
// Block tile 64(M) x 128(N), K-step 32. Double-buffered LDS tiles filled by
// global_load_async_to_lds_b128; s_wait_asynccnt + barrier pipeline.
#define BK 32
#define LDST 40  // bf16 per LDS row (80B: conflict-reducing, 16B-aligned)

__device__ inline void gemm_mainloop(const __bf16* __restrict__ A,
                                     const __bf16* __restrict__ BT,
                                     int K, int blockM, int blockN,
                                     __bf16 (*ldsA)[64 * LDST],
                                     __bf16 (*ldsB)[128 * LDST],
                                     f32x8 acc[2][2]) {
  const int tid = threadIdx.x;
  const int lane = tid & 31, w = tid >> 5;
  const int mi = w >> 2, ni = w & 3;
  const int lr = lane & 15, hi = lane >> 4;
  const int ldRow = tid >> 2;        // 0..63
  const int ldCol = (tid & 3) * 8;   // bf16 units, 16B chunk
  const __bf16* gA  = &A [(size_t)(blockM * 64 + ldRow) * K + ldCol];
  const __bf16* gB0 = &BT[(size_t)(blockN * 128 + ldRow) * K + ldCol];
  const __bf16* gB1 = &BT[(size_t)(blockN * 128 + 64 + ldRow) * K + ldCol];
  unsigned offA [2], offB0[2], offB1[2];
  for (int bb = 0; bb < 2; bb++) {
    offA [bb] = lds_off32(&ldsA[bb][ldRow * LDST + ldCol]);
    offB0[bb] = lds_off32(&ldsB[bb][ldRow * LDST + ldCol]);
    offB1[bb] = lds_off32(&ldsB[bb][(64 + ldRow) * LDST + ldCol]);
  }
  const int NK = K / BK;
  // prologue: fill buffer 0
  async_ld_b128(offA[0], gA);
  async_ld_b128(offB0[0], gB0);
  async_ld_b128(offB1[0], gB1);
  for (int kt = 0; kt < NK; kt++) {
    const int cur = kt & 1;
    wait_async0();        // this wave's copies into `cur` are done
    __syncthreads();      // everyone's copies done; everyone done reading 1-cur
    if (kt + 1 < NK) {    // prefetch next K-slab into the other buffer
      const int nxt = 1 - cur;
      const size_t ko = (size_t)(kt + 1) * BK;
      async_ld_b128(offA[nxt], gA + ko);
      async_ld_b128(offB0[nxt], gB0 + ko);
      async_ld_b128(offB1[nxt], gB1 + ko);
    }
    Frag a[2], b[2];
    for (int t = 0; t < 2; t++) {
      // A 16x32 frag: lanes 0-15 K{0..7,16..23}, lanes 16-31 K{8..15,24..31}
      const __bf16* ap = &ldsA[cur][(mi * 32 + t * 16 + lr) * LDST];
      a[t].h[0] = *(const bf16x8*)&ap[hi * 8];
      a[t].h[1] = *(const bf16x8*)&ap[16 + hi * 8];
      // B 32x16 frag: lane = column; lanes 0-15 K{0..7,16..23}, lanes 16-31 K{8..15,24..31}
      const __bf16* bp = &ldsB[cur][(ni * 32 + t * 16 + lr) * LDST];
      b[t].h[0] = *(const bf16x8*)&bp[hi * 8];
      b[t].h[1] = *(const bf16x8*)&bp[16 + hi * 8];
    }
    for (int ai = 0; ai < 2; ai++)
      for (int bi = 0; bi < 2; bi++)
        acc[ai][bi] = wmma_bf16(a[ai].v, b[bi].v, acc[ai][bi]);
  }
}

// QKV projection: C = h @ W + b, written head-split [b][h][s][d] as bf16
__global__ void __launch_bounds__(256)
gemm_qkv_kernel(const __bf16* __restrict__ A, const __bf16* __restrict__ WT,
                const float* __restrict__ bias, __bf16* __restrict__ out) {
  __shared__ __bf16 ldsA[2][64 * LDST];
  __shared__ __bf16 ldsB[2][128 * LDST];
  f32x8 acc[2][2] = {};
  gemm_mainloop(A, WT, D_MODEL, blockIdx.y, blockIdx.x, ldsA, ldsB, acc);
  const int tid = threadIdx.x, lane = tid & 31, w = tid >> 5;
  const int mi = w >> 2, ni = w & 3, hi = lane >> 4, lc = lane & 15;
  for (int ai = 0; ai < 2; ai++)
    for (int bi = 0; bi < 2; bi++) {
      int n_global = blockIdx.x * 128 + ni * 32 + bi * 16 + lc;
      float bv = bias[n_global];
      int h = n_global >> 6, d = n_global & 63;
      for (int v = 0; v < 8; v++) {
        int m_global = blockIdx.y * 64 + mi * 32 + ai * 16 + v + hi * 8;
        int b = m_global >> 11, s = m_global & (SEQ - 1);
        out[(((size_t)(b * N_HEADS + h)) * SEQ + s) * HEAD_DIM + d] =
            f2bf(acc[ai][bi][v] + bv);
      }
    }
}

// Output projection: out = (AO @ Wo + bo) * gate, fp32
__global__ void __launch_bounds__(256)
gemm_out_kernel(const __bf16* __restrict__ A, const __bf16* __restrict__ WT,
                const float* __restrict__ bias, const float* __restrict__ gate,
                float* __restrict__ out) {
  __shared__ __bf16 ldsA[2][64 * LDST];
  __shared__ __bf16 ldsB[2][128 * LDST];
  f32x8 acc[2][2] = {};
  gemm_mainloop(A, WT, D_MODEL, blockIdx.y, blockIdx.x, ldsA, ldsB, acc);
  const int tid = threadIdx.x, lane = tid & 31, w = tid >> 5;
  const int mi = w >> 2, ni = w & 3, hi = lane >> 4, lc = lane & 15;
  for (int ai = 0; ai < 2; ai++)
    for (int bi = 0; bi < 2; bi++) {
      int n_global = blockIdx.x * 128 + ni * 32 + bi * 16 + lc;
      float bv = bias[n_global];
      for (int v = 0; v < 8; v++) {
        int m_global = blockIdx.y * 64 + mi * 32 + ai * 16 + v + hi * 8;
        out[(size_t)m_global * D_MODEL + n_global] =
            (acc[ai][bi][v] + bv) * gate[m_global];
      }
    }
}

// ---------------------------------------------------------------- attention
// One block per (batch, head, 16-query strip). 256 threads = 8 waves.
// Phase 1: WMMA QK^T -> 16x2048 fp32 score strip resident in LDS (causal).
// Phase 2: exact per-row top-64 threshold via 4-round radix select (LDS hist).
// Phase 3: sparse softmax + A.V gather, one wave per row, packed b32 V loads.
__global__ void __launch_bounds__(256)
attn_kernel(const __bf16* __restrict__ Qb, const __bf16* __restrict__ Kb,
            const __bf16* __restrict__ Vb, __bf16* __restrict__ AO) {
  extern __shared__ float smem[];
  const int SCPAD = SEQ + 4;
  float*    sc     = smem;                             // 16 * SCPAD
  unsigned* hist   = (unsigned*)(smem + 16 * SCPAD);   // 256
  unsigned* tmp    = hist + 256;                       // 256
  float*    rowmax = (float*)(tmp + 256);              // 16
  float*    thresh = rowmax + 16;                      // 16
  __shared__ unsigned sb, skk;

  const int tid = threadIdx.x, lane = tid & 31, w = tid >> 5;
  const int bid = blockIdx.x;
  const int qt = bid & (SEQ / 16 - 1);
  const int bh = bid >> 7;
  const int h = bh & (N_HEADS - 1);
  const int b = bh >> 4;
  const int q0 = qt * 16;
  const int lr = lane & 15, hi = lane >> 4;

  // ---------------- Phase 1: scores via WMMA ----------------
  const __bf16* Qp  = Qb + (((size_t)(b * N_HEADS + h)) * SEQ + q0) * HEAD_DIM;
  const __bf16* Kp0 = Kb + ((size_t)(b * N_HEADS + h)) * SEQ * HEAD_DIM;
  Frag a0, a1;
  a0.h[0] = *(const bf16x8*)&Qp[lr * 64 + hi * 8];
  a0.h[1] = *(const bf16x8*)&Qp[lr * 64 + 16 + hi * 8];
  a1.h[0] = *(const bf16x8*)&Qp[lr * 64 + 32 + hi * 8];
  a1.h[1] = *(const bf16x8*)&Qp[lr * 64 + 48 + hi * 8];
  for (int kt = w; kt <= qt; kt += 8) {
    const __bf16* kp = Kp0 + ((size_t)kt * 16 + lr) * HEAD_DIM;
    Frag b0, b1;
    b0.h[0] = *(const bf16x8*)&kp[hi * 8];
    b0.h[1] = *(const bf16x8*)&kp[16 + hi * 8];
    b1.h[0] = *(const bf16x8*)&kp[32 + hi * 8];
    b1.h[1] = *(const bf16x8*)&kp[48 + hi * 8];
    f32x8 acc = {0.f,0.f,0.f,0.f,0.f,0.f,0.f,0.f};
    acc = wmma_bf16(a0.v, b0.v, acc);
    acc = wmma_bf16(a1.v, b1.v, acc);
    int key = kt * 16 + lr;
    for (int v = 0; v < 8; v++) {
      int m = v + hi * 8;
      int q = q0 + m;
      float val = (key <= q) ? acc[v] * 0.125f : -INFINITY;  // 1/sqrt(64)
      sc[m * SCPAD + key] = val;
    }
  }
  __syncthreads();

  // ---------------- Phase 2: row max + exact top-64 threshold ----------------
  for (int r = 0; r < 16; r++) {
    const int L = q0 + r + 1;
    float lm = -INFINITY;
    for (int i = tid; i < L; i += 256) lm = fmaxf(lm, sc[r * SCPAD + i]);
    float* redf = (float*)tmp;
    redf[tid] = lm;
    __syncthreads();
    for (int off = 128; off > 0; off >>= 1) {
      if (tid < off) redf[tid] = fmaxf(redf[tid], redf[tid + off]);
      __syncthreads();
    }
    if (tid == 0) rowmax[r] = redf[0];
    __syncthreads();
    if (L <= TOP_K) {
      if (tid == 0) thresh[r] = -INFINITY;
      __syncthreads();
      continue;
    }
    unsigned prefix = 0, kk = TOP_K;
    for (int round = 0; round < 4; round++) {
      const int shift = 24 - 8 * round;
      const unsigned maskHigh = (round == 0) ? 0u : (0xFFFFFFFFu << (shift + 8));
      hist[tid] = 0;
      __syncthreads();
      for (int i = tid; i < L; i += 256) {
        unsigned u = f2ord(sc[r * SCPAD + i]);
        if ((u & maskHigh) == (prefix & maskHigh))
          atomicAdd(&hist[(u >> shift) & 0xFF], 1u);
      }
      __syncthreads();
      tmp[tid] = hist[tid];
      __syncthreads();
      for (int off = 1; off < 256; off <<= 1) {
        unsigned v2 = tmp[tid] + ((tid + off < 256) ? tmp[tid + off] : 0u);
        __syncthreads();
        tmp[tid] = v2;
        __syncthreads();
      }
      {
        unsigned s0 = tmp[tid];
        unsigned s1 = (tid < 255) ? tmp[tid + 1] : 0u;
        if (s0 >= kk && s1 < kk) { sb = (unsigned)tid; skk = kk - s1; }
      }
      __syncthreads();
      prefix |= sb << shift;
      kk = skk;
      __syncthreads();
    }
    if (tid == 0) thresh[r] = ord2f(prefix);
    __syncthreads();
  }
  __syncthreads();

  // ---------------- Phase 3: sparse softmax + A.V (wave per row) ----------------
  const __bf16* Vp = Vb + ((size_t)(b * N_HEADS + h)) * SEQ * HEAD_DIM;
  for (int rr = 0; rr < 2; rr++) {
    const int r = w + rr * 8;
    const int q = q0 + r;
    const int L = q + 1;
    const float th = thresh[r], mx = rowmax[r];
    float es = 0.f;
    for (int i = lane; i < L; i += 32) {
      float s = sc[r * SCPAD + i];
      if (s >= th) es += __expf(s - mx);
    }
    for (int off = 16; off > 0; off >>= 1) es += __shfl_xor(es, off, 32);
    const float inv = 1.f / es;
    float o0 = 0.f, o1 = 0.f;
    const int dp = lane * 2;                 // lane owns dims dp, dp+1
    for (int i = 0; i < L; i++) {
      float s = sc[r * SCPAD + i];           // LDS broadcast, wave-coherent branch
      if (s >= th) {
        float p = __expf(s - mx) * inv;
        unsigned uv = *(const unsigned*)&Vp[(size_t)i * HEAD_DIM + dp];
        o0 += p * bits2f(uv << 16);          // low bf16
        o1 += p * bits2f(uv & 0xFFFF0000u);  // high bf16
      }
    }
    size_t row = (size_t)(b * SEQ + q) * D_MODEL + (size_t)h * HEAD_DIM;
    unsigned packed = (unsigned)f2bf_bits(o0) | ((unsigned)f2bf_bits(o1) << 16);
    *(unsigned*)&AO[row + dp] = packed;
  }
}

// ---------------------------------------------------------------- launch
extern "C" void kernel_launch(void* const* d_in, const int* in_sizes, int n_in,
                              void* d_out, int out_size, void* d_ws, size_t ws_size,
                              hipStream_t stream) {
  const float* h_ssm = (const float*)d_in[0];
  const float* gate  = (const float*)d_in[1];
  const float* Wq = (const float*)d_in[2]; const float* bq = (const float*)d_in[3];
  const float* Wk = (const float*)d_in[4]; const float* bk = (const float*)d_in[5];
  const float* Wv = (const float*)d_in[6]; const float* bv = (const float*)d_in[7];
  const float* Wo = (const float*)d_in[8]; const float* bo = (const float*)d_in[9];
  float* out = (float*)d_out;

  char* ws = (char*)d_ws;
  size_t off = 0;
  auto alloc = [&](size_t bytes) -> char* {
    char* p = ws + off; off += (bytes + 255) & ~(size_t)255; return p;
  };
  __bf16* hB  = (__bf16*)alloc((size_t)M_ROWS * D_MODEL * 2);
  __bf16* WqT = (__bf16*)alloc((size_t)D_MODEL * D_MODEL * 2);
  __bf16* WkT = (__bf16*)alloc((size_t)D_MODEL * D_MODEL * 2);
  __bf16* WvT = (__bf16*)alloc((size_t)D_MODEL * D_MODEL * 2);
  __bf16* WoT = (__bf16*)alloc((size_t)D_MODEL * D_MODEL * 2);
  __bf16* Qh  = (__bf16*)alloc((size_t)M_ROWS * D_MODEL * 2);
  __bf16* Kh  = (__bf16*)alloc((size_t)M_ROWS * D_MODEL * 2);
  __bf16* Vh  = (__bf16*)alloc((size_t)M_ROWS * D_MODEL * 2);
  __bf16* AO  = (__bf16*)alloc((size_t)M_ROWS * D_MODEL * 2);

  cvt_f32_bf16_kernel<<<(M_ROWS * D_MODEL) / 256, 256, 0, stream>>>(
      h_ssm, hB, M_ROWS * D_MODEL);

  dim3 tb(32, 8), tg(D_MODEL / 32, D_MODEL / 32);
  transpose_w_kernel<<<tg, tb, 0, stream>>>(Wq, WqT);
  transpose_w_kernel<<<tg, tb, 0, stream>>>(Wk, WkT);
  transpose_w_kernel<<<tg, tb, 0, stream>>>(Wv, WvT);
  transpose_w_kernel<<<tg, tb, 0, stream>>>(Wo, WoT);

  dim3 ggrid(D_MODEL / 128, M_ROWS / 64);  // (8, 64)
  gemm_qkv_kernel<<<ggrid, 256, 0, stream>>>(hB, WqT, bq, Qh);
  gemm_qkv_kernel<<<ggrid, 256, 0, stream>>>(hB, WkT, bk, Kh);
  gemm_qkv_kernel<<<ggrid, 256, 0, stream>>>(hB, WvT, bv, Vh);

  size_t smem = (size_t)(16 * (SEQ + 4) + 256 + 256 + 16 + 16) * sizeof(float);
  (void)hipFuncSetAttribute((const void*)attn_kernel,
                            hipFuncAttributeMaxDynamicSharedMemorySize, (int)smem);
  attn_kernel<<<BATCH * N_HEADS * (SEQ / 16), 256, smem, stream>>>(Qh, Kh, Vh, AO);

  gemm_out_kernel<<<ggrid, 256, 0, stream>>>(AO, WoT, bo, gate, out);
}
